// Attention_8083128451525
// MI455X (gfx1250) — matrix operand order repaired and checked by reading, not compiled, not run on hardware
//
#include <hip/hip_runtime.h>
#include <hip/hip_bf16.h>

// ---------------------------------------------------------------------------
// Problem constants (fixed by the reference: B=8, H=W=32, dim=768, 12 heads)
// ---------------------------------------------------------------------------
#define B_N   8
#define H_N   32
#define W_N   32
#define L_N   1024          // H*W
#define DIM_N 768
#define NH_N  12
#define HD_N  64            // head dim
#define TD_N  2304          // 3*dim

typedef _Float16 half8 __attribute__((ext_vector_type(8)));
typedef _Float16 v16h  __attribute__((ext_vector_type(16)));
typedef float    v8f   __attribute__((ext_vector_type(8)));

__device__ __forceinline__ v8f vzero8() {
    v8f z;
#pragma unroll
    for (int i = 0; i < 8; ++i) z[i] = 0.0f;
    return z;
}

__device__ __forceinline__ v16h make_frag(half8 lo, half8 hi) {
    v16h r;
#pragma unroll
    for (int i = 0; i < 8; ++i) { r[i] = lo[i]; r[i + 8] = hi[i]; }
    return r;
}

// A fragment (16x32 f16, row-major source). rowp = this lane's row at K-offset 0.
// Lanes 0-15 hold row m=lane, K[0..7] + K[16..23]; lanes 16-31 hold K[8..15]+K[24..31].
__device__ __forceinline__ v16h load_a(const _Float16* rowp, int lane) {
    const int koff = (lane >> 4) * 8;
    half8 lo = *(const half8*)(rowp + koff);
    half8 hi = *(const half8*)(rowp + 16 + koff);
    return make_frag(lo, hi);
}

// B fragment (32x16 f16). Lane n<16 = column n, K[0..7] + K[16..23]; lane n+16 = K[8..15]+K[24..31].
// rowp = the memory row that supplies this lane's column, at K-offset 0.
__device__ __forceinline__ v16h load_b(const _Float16* rowp, int lane) {
    const int kbase = (lane >> 4) * 8;
    half8 lo = *(const half8*)(rowp + kbase);
    half8 hi = *(const half8*)(rowp + kbase + 16);
    return make_frag(lo, hi);
}

__device__ __forceinline__ v8f wmma32(v16h a, v16h b, v8f c) {
    return __builtin_amdgcn_wmma_f32_16x16x32_f16(false, a, false, b,
                                                  (short)0, c, false, false);
}

// ---------------------------------------------------------------------------
// Kernel 1: f32 -> f16 conversion (x, qkv_w, proj_w, rel_pos_h, rel_pos_w)
// ---------------------------------------------------------------------------
__global__ void cvt_f16_kernel(const float* __restrict__ src,
                               _Float16* __restrict__ dst, int n) {
    int i = blockIdx.x * blockDim.x + threadIdx.x;
    if (i < n) dst[i] = (_Float16)src[i];
}

// ---------------------------------------------------------------------------
// Kernel 2: QKV GEMM (8192x768 @ 768x2304^T) + bias + 2D RoPE, split q/k/v.
// One wave computes a 16(M tokens) x 64(N outputs) tile.
//   q -> qh[b][h][l][c] f16 (rope'd), k -> kh likewise, v -> vT[b][h][c][l] f16.
// ---------------------------------------------------------------------------
__global__ __launch_bounds__(128) void qkv_rope_kernel(
        const _Float16* __restrict__ xh, const _Float16* __restrict__ wh,
        const float* __restrict__ qkv_b,
        _Float16* __restrict__ qh, _Float16* __restrict__ kh,
        _Float16* __restrict__ vT) {
    const int lane = threadIdx.x & 31;
    const int wg   = blockIdx.x * (blockDim.x >> 5) + (threadIdx.x >> 5);
    const int mtile = wg / (TD_N / 64);
    const int n64   = wg % (TD_N / 64);
    const int row0  = mtile * 16;
    const int nlo   = lane & 15;
    const int halfw = lane >> 4;

    v8f acc[4];
#pragma unroll
    for (int j = 0; j < 4; ++j) acc[j] = vzero8();

    const _Float16* arow = xh + (size_t)(row0 + nlo) * DIM_N;
    for (int k0 = 0; k0 < DIM_N; k0 += 32) {
        __builtin_prefetch(arow + k0 + 96, 0, 1);
        v16h a = load_a(arow + k0, lane);
#pragma unroll
        for (int j = 0; j < 4; ++j) {
            const _Float16* brow =
                wh + (size_t)(n64 * 64 + j * 16 + nlo) * DIM_N + k0;
            acc[j] = wmma32(a, load_b(brow, lane), acc[j]);
        }
    }

    const int which = n64 / NH_N;   // 0=q, 1=k, 2=v (uniform per wave)
    const int head  = n64 % NH_N;
#pragma unroll
    for (int j = 0; j < 4; ++j) {
#pragma unroll
        for (int r = 0; r < 8; ++r) {
            const int m    = r + 8 * halfw;
            const int grow = row0 + m;              // global token row
            const int b    = grow >> 10;
            const int l    = grow & (L_N - 1);
            const int c    = j * 16 + nlo;          // channel within head
            const int d    = n64 * 64 + c;          // output column in [0,2304)
            float v = acc[j][r] + qkv_b[d];
            if (which == 2) {
                vT[(((size_t)(b * NH_N + head)) * HD_N + c) * L_N + l] = (_Float16)v;
            } else {
                // axial 2D RoPE: pair p=c/2; p<16 uses w*f[p], else h*f[p-16]
                const int w  = l & 31, hc = l >> 5;
                const int p  = c >> 1;
                const int fj = p & 15;
                const float coord = (p < 16) ? (float)w : (float)hc;
                const float ang = coord * __expf(-(float)fj * 0.5756462732485114f);
                const float cs = __cosf(ang), sn = __sinf(ang);
                const float partner = __shfl_xor(v, 1);   // channel c^1
                const float o = (c & 1) ? (partner * sn + v * cs)
                                        : (v * cs - partner * sn);
                _Float16* dst = (which == 0) ? qh : kh;
                dst[(((size_t)(b * NH_N + head)) * L_N + l) * HD_N + c] = (_Float16)o;
            }
        }
    }
}

// ---------------------------------------------------------------------------
// Kernel 3: decomposed relative-position bias.
//   mode 0: rel_h[b,n,l,kh] = q[l] . rel_pos_h[h(l)-kh+31]   (16 tokens, h fixed)
//   mode 1: rel_w[b,n,l,kw] = q[l] . rel_pos_w[w(l)-kw+31]   (16 tokens, w fixed)
// One wave -> one 16x32 output tile via 4 WMMAs.
// ---------------------------------------------------------------------------
__global__ __launch_bounds__(128) void relbias_kernel(
        const _Float16* __restrict__ qh,
        const _Float16* __restrict__ rhh, const _Float16* __restrict__ rwh,
        float* __restrict__ relh, float* __restrict__ relw) {
    const int lane = threadIdx.x & 31;
    int wg = blockIdx.x * (blockDim.x >> 5) + (threadIdx.x >> 5);
    const int mode = wg / (B_N * NH_N * 64);
    wg -= mode * (B_N * NH_N * 64);
    const int bh   = wg / 64;
    const int tile = wg % 64;
    const int nlo   = lane & 15;
    const int halfw = lane >> 4;

    const _Float16* rel = mode ? rwh : rhh;
    float* outp         = mode ? relw : relh;

    // this lane's A-row token and the fixed spatial coordinate of the tile
    int l_m, coord;
    if (mode == 0) { l_m = tile * 16 + nlo;                   coord = tile >> 1; }
    else { const int w = tile >> 1, h0 = (tile & 1) * 16;
           l_m = (h0 + nlo) * W_N + w;                        coord = w; }

    v8f s[2];
    s[0] = vzero8(); s[1] = vzero8();
    const _Float16* qrow = qh + ((size_t)bh * L_N + l_m) * HD_N;
#pragma unroll
    for (int k0 = 0; k0 < HD_N; k0 += 32) {
        v16h a = load_a(qrow + k0, lane);
#pragma unroll
        for (int j = 0; j < 2; ++j) {
            const int col = j * 16 + nlo;                     // relative key index
            const _Float16* brow = rel + (size_t)(coord - col + 31) * HD_N + k0;
            s[j] = wmma32(a, load_b(brow, lane), s[j]);
        }
    }
#pragma unroll
    for (int j = 0; j < 2; ++j)
#pragma unroll
        for (int r = 0; r < 8; ++r) {
            const int m = r + 8 * halfw;
            const int l = (mode == 0) ? (tile * 16 + m)
                                      : (((tile & 1) * 16 + m) * W_N + (tile >> 1));
            outp[((size_t)bh * L_N + l) * 32 + j * 16 + nlo] = s[j][r];
        }
}

// ---------------------------------------------------------------------------
// Kernel 4: fused flash attention. One wave owns a 16-row q tile for one
// (batch,head); streams 32-key blocks: S = QK^T/8 + rel_h + rel_w, online
// softmax, P (f16, via LDS C->A relayout) @ V^T accumulated in f32.
// ---------------------------------------------------------------------------
__global__ __launch_bounds__(128) void attn_kernel(
        const _Float16* __restrict__ qh, const _Float16* __restrict__ kh,
        const _Float16* __restrict__ vT,
        const float* __restrict__ relh, const float* __restrict__ relw,
        _Float16* __restrict__ aout) {
    __shared__ __align__(16) _Float16 lds[4][16 * 40];   // padded P tiles, per wave
    const int lane  = threadIdx.x & 31;
    const int wv    = threadIdx.x >> 5;
    const int wg    = blockIdx.x * 4 + wv;
    const int bh    = wg / 64;            // b*12 + head
    const int qtile = wg % 64;
    const int q0    = qtile * 16;
    const int nlo   = lane & 15;
    const int halfw = lane >> 4;

    const _Float16* qrow = qh + ((size_t)bh * L_N + q0 + nlo) * HD_N;
    const v16h qa0 = load_a(qrow, lane);
    const v16h qa1 = load_a(qrow + 32, lane);

    v8f acc[4];
#pragma unroll
    for (int j = 0; j < 4; ++j) acc[j] = vzero8();
    float rmax[8], rsum[8];
#pragma unroll
    for (int r = 0; r < 8; ++r) { rmax[r] = -1e30f; rsum[r] = 0.0f; }

    const float* rhp = relh + (size_t)bh * L_N * 32;
    const float* rwp = relw + (size_t)bh * L_N * 32;
    _Float16* pl = lds[wv];

    for (int kb = 0; kb < L_N; kb += 32) {
        // ---- scores: two 16x16 tiles, K-depth 64 ----
        v8f s[2];
#pragma unroll
        for (int j = 0; j < 2; ++j) {
            const _Float16* krow =
                kh + ((size_t)bh * L_N + kb + j * 16 + nlo) * HD_N;
            v8f c = vzero8();
            c = wmma32(qa0, load_b(krow, lane), c);
            c = wmma32(qa1, load_b(krow + 32, lane), c);
            s[j] = c;
        }
        // ---- scale + rel bias, tile row-max ----
        float pv[2][8], tmax[8];
#pragma unroll
        for (int r = 0; r < 8; ++r) {
            const int l = q0 + r + 8 * halfw;
#pragma unroll
            for (int j = 0; j < 2; ++j) {
                const int key = kb + j * 16 + nlo;
                pv[j][r] = s[j][r] * 0.125f
                         + rhp[(size_t)l * 32 + (key >> 5)]
                         + rwp[(size_t)l * 32 + (key & 31)];
            }
            float t = fmaxf(pv[0][r], pv[1][r]);
#pragma unroll
            for (int msk = 1; msk < 16; msk <<= 1) t = fmaxf(t, __shfl_xor(t, msk));
            tmax[r] = t;
        }
        // ---- online softmax update ----
#pragma unroll
        for (int r = 0; r < 8; ++r) {
            const float mnew  = fmaxf(rmax[r], tmax[r]);
            const float scale = __expf(rmax[r] - mnew);
            rmax[r] = mnew;
            const float p0 = __expf(pv[0][r] - mnew);
            const float p1 = __expf(pv[1][r] - mnew);
            pv[0][r] = p0; pv[1][r] = p1;
            float ts = p0 + p1;
#pragma unroll
            for (int msk = 1; msk < 16; msk <<= 1) ts += __shfl_xor(ts, msk);
            rsum[r] = rsum[r] * scale + ts;
#pragma unroll
            for (int jo = 0; jo < 4; ++jo) acc[jo][r] *= scale;
        }
        // ---- P: C-layout -> A-layout through LDS (f16) ----
#pragma unroll
        for (int j = 0; j < 2; ++j)
#pragma unroll
            for (int r = 0; r < 8; ++r)
                pl[(r + 8 * halfw) * 40 + j * 16 + nlo] = (_Float16)pv[j][r];
        __builtin_amdgcn_wave_barrier();
        const v16h pfrag = load_a(pl + nlo * 40, lane);
        __builtin_amdgcn_wave_barrier();
        // ---- acc += P @ V  (V stored transposed: contiguous per-lane rows) ----
#pragma unroll
        for (int jo = 0; jo < 4; ++jo) {
            const _Float16* vrow =
                vT + ((size_t)bh * HD_N + jo * 16 + nlo) * L_N + kb;
            acc[jo] = wmma32(pfrag, load_b(vrow, lane), acc[jo]);
        }
    }
    // ---- normalize, store f16 (B,NH,L,HD) for the proj GEMM ----
#pragma unroll
    for (int jo = 0; jo < 4; ++jo)
#pragma unroll
        for (int r = 0; r < 8; ++r) {
            const int m = r + 8 * halfw;
            const float o = acc[jo][r] / rsum[r];
            aout[((size_t)bh * L_N + q0 + m) * HD_N + jo * 16 + nlo] = (_Float16)o;
        }
}

// ---------------------------------------------------------------------------
// Kernel 5: output projection (8192x768 @ 768x768^T) + bias, f32 output.
// A is the head-interleaved attention output; per K-step the 32 channels stay
// inside one head, so per-lane loads remain contiguous.
// ---------------------------------------------------------------------------
__global__ __launch_bounds__(128) void proj_kernel(
        const _Float16* __restrict__ ah, const _Float16* __restrict__ pwh,
        const float* __restrict__ proj_b, float* __restrict__ out) {
    const int lane = threadIdx.x & 31;
    const int wg   = blockIdx.x * (blockDim.x >> 5) + (threadIdx.x >> 5);
    const int mtile = wg / (DIM_N / 64);
    const int n64   = wg % (DIM_N / 64);
    const int row0  = mtile * 16;
    const int nlo   = lane & 15;
    const int halfw = lane >> 4;

    const int grow_a = row0 + nlo;
    const int ba = grow_a >> 10;
    const int la = grow_a & (L_N - 1);

    v8f acc[4];
#pragma unroll
    for (int j = 0; j < 4; ++j) acc[j] = vzero8();

    for (int k0 = 0; k0 < DIM_N; k0 += 32) {
        const int head = k0 >> 6, c0 = k0 & 63;
        const _Float16* arow =
            ah + (((size_t)(ba * NH_N + head)) * L_N + la) * HD_N + c0;
        __builtin_prefetch(arow + 32, 0, 1);
        v16h a = load_a(arow, lane);
#pragma unroll
        for (int j = 0; j < 4; ++j) {
            const _Float16* brow =
                pwh + (size_t)(n64 * 64 + j * 16 + nlo) * DIM_N + k0;
            acc[j] = wmma32(a, load_b(brow, lane), acc[j]);
        }
    }
#pragma unroll
    for (int j = 0; j < 4; ++j)
#pragma unroll
        for (int r = 0; r < 8; ++r) {
            const int d    = n64 * 64 + j * 16 + nlo;
            const int grow = row0 + r + 8 * halfw;
            out[(size_t)grow * DIM_N + d] = acc[j][r] + proj_b[d];
        }
}

// ---------------------------------------------------------------------------
// Launch
// ---------------------------------------------------------------------------
extern "C" void kernel_launch(void* const* d_in, const int* in_sizes, int n_in,
                              void* d_out, int out_size, void* d_ws, size_t ws_size,
                              hipStream_t stream) {
    const float* x     = (const float*)d_in[0];
    const float* qkv_w = (const float*)d_in[1];
    const float* qkv_b = (const float*)d_in[2];
    const float* pro_w = (const float*)d_in[3];
    const float* pro_b = (const float*)d_in[4];
    const float* rp_h  = (const float*)d_in[5];
    const float* rp_w  = (const float*)d_in[6];
    float* out = (float*)d_out;

    char* ws = (char*)d_ws;
    size_t off = 0;
    auto take = [&](size_t bytes) {
        char* p = ws + off;
        off += (bytes + 255) & ~(size_t)255;
        return p;
    };
    _Float16* xh  = (_Float16*)take((size_t)B_N * L_N * DIM_N * 2);
    _Float16* wh  = (_Float16*)take((size_t)TD_N * DIM_N * 2);
    _Float16* pwh = (_Float16*)take((size_t)DIM_N * DIM_N * 2);
    _Float16* rhh = (_Float16*)take((size_t)(2 * H_N - 1) * HD_N * 2);
    _Float16* rwh = (_Float16*)take((size_t)(2 * W_N - 1) * HD_N * 2);
    _Float16* qh  = (_Float16*)take((size_t)B_N * NH_N * L_N * HD_N * 2);
    _Float16* kh  = (_Float16*)take((size_t)B_N * NH_N * L_N * HD_N * 2);
    _Float16* vT  = (_Float16*)take((size_t)B_N * NH_N * HD_N * L_N * 2);
    float*    relh = (float*)take((size_t)B_N * NH_N * L_N * 32 * 4);
    float*    relw = (float*)take((size_t)B_N * NH_N * L_N * 32 * 4);
    _Float16* ah  = (_Float16*)take((size_t)B_N * NH_N * L_N * HD_N * 2);
    (void)ws_size;

    auto cvt = [&](const float* s, _Float16* d, int n) {
        cvt_f16_kernel<<<(n + 255) / 256, 256, 0, stream>>>(s, d, n);
    };
    cvt(x,     xh,  B_N * L_N * DIM_N);
    cvt(qkv_w, wh,  TD_N * DIM_N);
    cvt(pro_w, pwh, DIM_N * DIM_N);
    cvt(rp_h,  rhh, (2 * H_N - 1) * HD_N);
    cvt(rp_w,  rwh, (2 * W_N - 1) * HD_N);

    // QKV + RoPE: 512 M-tiles x 36 N64-tiles = 18432 waves, 4 waves/block
    qkv_rope_kernel<<<4608, 128, 0, stream>>>(xh, wh, qkv_b, qh, kh, vT);
    // rel bias: 2 modes x 96 (b,head) x 64 tiles = 12288 waves
    relbias_kernel<<<3072, 128, 0, stream>>>(qh, rhh, rwh, relh, relw);
    // attention: 96 x 64 q-tiles = 6144 waves
    attn_kernel<<<1536, 128, 0, stream>>>(qh, kh, vT, relh, relw, ah);
    // projection: 512 M-tiles x 12 N64-tiles = 6144 waves
    proj_kernel<<<1536, 128, 0, stream>>>(ah, pwh, pro_b, out);
}
